// AdjMultiHeadAttention_41738492182541
// MI455X (gfx1250) — matrix operand order repaired and checked by reading, not compiled, not run on hardware
//
#include <hip/hip_runtime.h>
#include <hip/hip_bf16.h>

#define EMBED 512
#define HEADS 8
#define HD    64
#define NB    16
#define SEQ   1024
#define ATT_SCALE 0.125f   // 64^-0.5

typedef __bf16 bf16_t;
typedef __attribute__((ext_vector_type(16))) __bf16 v16bf;
typedef __attribute__((ext_vector_type(4)))  __bf16 v4bf;
typedef __attribute__((ext_vector_type(8)))  float  v8f;
typedef __attribute__((ext_vector_type(4)))  unsigned int u32x4;
typedef __attribute__((ext_vector_type(8)))  int i32x8;
typedef __attribute__((ext_vector_type(4)))  int i32x4;

#if defined(__has_builtin)
#  if __has_builtin(__builtin_amdgcn_tensor_load_to_lds) && __has_builtin(__builtin_amdgcn_s_wait_tensorcnt)
#    define HAVE_TDM 1
#  endif
#endif
#ifndef HAVE_TDM
#  define HAVE_TDM 0
#endif

// ISA 7.12.2: bf16 A-matrix 16xK; element e (0..15) -> K = e + 8*laneHi + 8*(e>=8).
__device__ __forceinline__ int a_koff(int e, int hi) { return e + 8 * hi + ((e >> 3) << 3); }
// bf16 B-matrix 32x16: same K order as A; element e -> K = e + 8*laneHi + 8*(e>=8).
__device__ __forceinline__ int b_koff(int e, int hi) { return e + 8 * hi + ((e >> 3) << 3); }

#if HAVE_TDM
// TDM: DMA one contiguous 4 KB chunk (512 x 8B) from global into LDS.
// D# per ISA ch.8: group0 = {count=1, lds_addr, global_addr[56:0], type=2};
// group1 = {data_size=3(8B), tensor_dim0=512, tensor_dim1=1, tile_dim0=512,
//           tile_dim1=1, tensor_dim0_stride=512}.
__device__ __forceinline__ void tdm_load_4kb(unsigned lds_addr, unsigned long long gaddr)
{
    u32x4 g0;
    g0[0] = 1u;  // count = 1 valid descriptor
    g0[1] = (unsigned)__builtin_amdgcn_readfirstlane((int)lds_addr);
    g0[2] = (unsigned)__builtin_amdgcn_readfirstlane((int)(unsigned)gaddr);
    g0[3] = ((unsigned)__builtin_amdgcn_readfirstlane((int)(unsigned)(gaddr >> 32)) & 0x1FFFFFFu)
            | 0x80000000u;  // type=2 ("image") in bits 127:126
    i32x8 g1 = { (int)0x00030000,        // data_size = 3 (8B), workgroup_mask = 0
                 (int)(512u << 16),      // tensor_dim0 = 512 (bits 79:48, low half)
                 (int)(1u << 16),        // tensor_dim1 = 1   (bits 111:80, low half)
                 (int)(512u << 16),      // tile_dim0 = 512   (bits 127:112)
                 1,                      // tile_dim1 = 1
                 512,                    // tensor_dim0_stride = 512
                 0, 0 };
    i32x4 z4 = {0, 0, 0, 0};
#if __clang_major__ >= 23
    i32x8 z8 = {0, 0, 0, 0, 0, 0, 0, 0};
    __builtin_amdgcn_tensor_load_to_lds(g0, g1, z4, z4, z8, 0);
#else
    __builtin_amdgcn_tensor_load_to_lds(g0, g1, z4, z4, 0);
#endif
}
#endif

// ---------------------------------------------------------------------------
// Kernel 0: one-shot weight conversion f32 -> bf16 (Wq,Wk,Wv,Wo consecutive).
// ---------------------------------------------------------------------------
__global__ void __launch_bounds__(256)
wcvt_kernel(const float* __restrict__ Wq, const float* __restrict__ Wk,
            const float* __restrict__ Wv, const float* __restrict__ Wo,
            bf16_t* __restrict__ out)
{
    int idx = blockIdx.x * 256 + threadIdx.x;        // 0 .. 262143 (float4 units)
    int m = idx >> 16, off = idx & 65535;            // 65536 float4 per matrix
    const float4* src = (const float4*)(m == 0 ? Wq : m == 1 ? Wk : m == 2 ? Wv : Wo);
    float4 t = src[off];
    v4bf o; o[0] = (__bf16)t.x; o[1] = (__bf16)t.y; o[2] = (__bf16)t.z; o[3] = (__bf16)t.w;
    ((v4bf*)out)[idx] = o;
}

// ---------------------------------------------------------------------------
// Kernel 1: fused Q/K/V projection.  Y[m,f] = sum_e x[m,e] * W[f,e] + b[f]
// Output layout: [B, H, S, D] bf16.
// ---------------------------------------------------------------------------
__global__ void __launch_bounds__(256)
qkv_proj_kernel(const float* __restrict__ x, const bf16_t* __restrict__ Wb,
                const float* __restrict__ bq, const float* __restrict__ bk,
                const float* __restrict__ bv,
                bf16_t* __restrict__ Qo, bf16_t* __restrict__ Ko, bf16_t* __restrict__ Vo)
{
    __shared__ __align__(16) bf16_t xs[16 * EMBED];   // 16 KB: x tile in bf16
    const int tid  = threadIdx.x;
    const int lane = tid & 31, wave = tid >> 5;
    const int n = lane & 15, hi = lane >> 4;
    const int mt = blockIdx.x;
    const int nt = blockIdx.y * 8 + wave;

    const bf16_t* W; const float* bias; bf16_t* out;
    if      (blockIdx.z == 0) { W = Wb;               bias = bq; out = Qo; }
    else if (blockIdx.z == 1) { W = Wb + 262144;      bias = bk; out = Ko; }
    else                      { W = Wb + 2 * 262144;  bias = bv; out = Vo; }

    // stage + convert the x tile once (shared by all 8 waves)
    const float4* src = (const float4*)(x + (size_t)mt * 16 * EMBED);
    v4bf* dst = (v4bf*)xs;
    #pragma unroll
    for (int i = 0; i < 8; ++i) {
        float4 t = src[tid + 256 * i];
        v4bf o; o[0] = (__bf16)t.x; o[1] = (__bf16)t.y; o[2] = (__bf16)t.z; o[3] = (__bf16)t.w;
        dst[tid + 256 * i] = o;
    }
    __syncthreads();

    const int f0 = nt * 16;
    const bf16_t* wrow = W + (size_t)(f0 + n) * EMBED;   // B[e, n] = W[f0+n, e]
    v8f acc = {};
    #pragma unroll
    for (int e0 = 0; e0 < EMBED; e0 += 32) {
        v16bf a, bm;
        #pragma unroll
        for (int e = 0; e < 16; ++e) {
            a[e]  = xs[n * EMBED + e0 + a_koff(e, hi)];
            bm[e] = wrow[e0 + b_koff(e, hi)];
        }
        acc = __builtin_amdgcn_wmma_f32_16x16x32_bf16(false, a, false, bm,
                                                      (short)0, acc, false, false);
    }
    const float bia = bias[f0 + n];
    const int h = (f0 + n) >> 6, d = (f0 + n) & 63;
    #pragma unroll
    for (int r = 0; r < 8; ++r) {
        int m  = mt * 16 + r + 8 * hi;
        int bb = m >> 10, s = m & (SEQ - 1);
        out[(((size_t)bb * HEADS + h) * SEQ + s) * HD + d] = (__bf16)(acc[r] + bia);
    }
}

// ---------------------------------------------------------------------------
// Kernel 2: fused masked attention, flash-style online softmax.
// Block = 8 waves = 8 heads on the same 16 q-rows; mask tile shared via LDS.
// K/V chunks staged per-wave by the Tensor Data Mover, double-buffered.
// ---------------------------------------------------------------------------
__global__ void __launch_bounds__(256)
attn_kernel(const bf16_t* __restrict__ Q, const bf16_t* __restrict__ K,
            const bf16_t* __restrict__ V,
            const float* __restrict__ adj, const float* __restrict__ bond,
            bf16_t* __restrict__ ctx)
{
    __shared__ __align__(16) bf16_t ks[8][2][32 * HD];  // per-wave K, 2 buffers
    __shared__ __align__(16) bf16_t vs[8][2][32 * HD];  // per-wave V, 2 buffers
    __shared__ __align__(16) bf16_t ps[8][16 * 32];     // per-wave P relayout
    __shared__ __align__(16) float  msk[16 * 32];       // mask tile, head-shared

    const int tid  = threadIdx.x;
    const int lane = tid & 31, wave = tid >> 5;
    const int n = lane & 15, hi = lane >> 4;
    const int b  = blockIdx.y;
    const int h  = wave;
    const int q0 = blockIdx.x * 16;

    const bf16_t* Qb = Q + ((size_t)b * HEADS + h) * SEQ * HD;
    const bf16_t* Kb = K + ((size_t)b * HEADS + h) * SEQ * HD;
    const bf16_t* Vb = V + ((size_t)b * HEADS + h) * SEQ * HD;
    const float* adj_b  = adj  + (size_t)b * SEQ * SEQ;
    const float* bond_b = bond + (size_t)b * SEQ * SEQ;

    // Q tile (16 rows x 64) as two A fragments
    v16bf qa0, qa1;
    #pragma unroll
    for (int e = 0; e < 16; ++e) {
        qa0[e] = Qb[(size_t)(q0 + n) * HD +      a_koff(e, hi)];
        qa1[e] = Qb[(size_t)(q0 + n) * HD + 32 + a_koff(e, hi)];
    }

    v8f o0 = {}, o1 = {}, o2 = {}, o3 = {};
    float mrow[8], lrow[8];
    #pragma unroll
    for (int r = 0; r < 8; ++r) { mrow[r] = -3.0e38f; lrow[r] = 0.f; }

#if HAVE_TDM
    const unsigned ldsK = (unsigned)(size_t)&ks[wave][0][0];
    const unsigned ldsV = (unsigned)(size_t)&vs[wave][0][0];
    // prime buffer 0 with chunk 0
    tdm_load_4kb(ldsK, (unsigned long long)(size_t)Kb);
    tdm_load_4kb(ldsV, (unsigned long long)(size_t)Vb);
#endif
    int buf = 0;

    for (int kb0 = 0; kb0 < SEQ; kb0 += 32) {
#if HAVE_TDM
        if (kb0 + 32 < SEQ) {   // issue next chunk's DMA into the other buffer
            tdm_load_4kb(ldsK + (buf ^ 1) * 4096,
                         (unsigned long long)(size_t)(Kb + (size_t)(kb0 + 32) * HD));
            tdm_load_4kb(ldsV + (buf ^ 1) * 4096,
                         (unsigned long long)(size_t)(Vb + (size_t)(kb0 + 32) * HD));
            __builtin_amdgcn_s_wait_tensorcnt(2);   // current buffer's 2 loads done
        } else {
            __builtin_amdgcn_s_wait_tensorcnt(0);
        }
#else
        {
            const uint4* kg = (const uint4*)(Kb + (size_t)kb0 * HD);
            const uint4* vg = (const uint4*)(Vb + (size_t)kb0 * HD);
            uint4* kd = (uint4*)ks[wave][0];
            uint4* vd = (uint4*)vs[wave][0];
            #pragma unroll
            for (int i = 0; i < 8; ++i) {
                kd[lane + 32 * i] = kg[lane + 32 * i];
                vd[lane + 32 * i] = vg[lane + 32 * i];
            }
        }
#endif
        const bf16_t* ksw = ks[wave][buf];
        const bf16_t* vsw = vs[wave][buf];

        // stage mask tile once per block (shared by all 8 head-waves)
        for (int i = tid; i < 512; i += 256) {
            int qr = i >> 5, kk = i & 31;
            size_t off = (size_t)(q0 + qr) * SEQ + kb0 + kk;
            msk[i] = adj_b[off] + bond_b[off];
            if (kb0 + 32 < SEQ) {
                __builtin_prefetch(adj_b + off + 32, 0, 1);
                __builtin_prefetch(bond_b + off + 32, 0, 1);
            }
        }
        __syncthreads();

        // S = Q K^T : two 16x16 key tiles, head-dim 64 = 2 WMMAs each
        v8f s0 = {}, s1 = {};
        {
            v16bf bm;
            #pragma unroll
            for (int e = 0; e < 16; ++e) bm[e] = ksw[n * HD + b_koff(e, hi)];
            s0 = __builtin_amdgcn_wmma_f32_16x16x32_bf16(false, qa0, false, bm, (short)0, s0, false, false);
            #pragma unroll
            for (int e = 0; e < 16; ++e) bm[e] = ksw[n * HD + 32 + b_koff(e, hi)];
            s0 = __builtin_amdgcn_wmma_f32_16x16x32_bf16(false, qa1, false, bm, (short)0, s0, false, false);
            #pragma unroll
            for (int e = 0; e < 16; ++e) bm[e] = ksw[(16 + n) * HD + b_koff(e, hi)];
            s1 = __builtin_amdgcn_wmma_f32_16x16x32_bf16(false, qa0, false, bm, (short)0, s1, false, false);
            #pragma unroll
            for (int e = 0; e < 16; ++e) bm[e] = ksw[(16 + n) * HD + 32 + b_koff(e, hi)];
            s1 = __builtin_amdgcn_wmma_f32_16x16x32_bf16(false, qa1, false, bm, (short)0, s1, false, false);
        }

        // scale * mask, online softmax update (row M = r + 8*hi, col N = n)
        #pragma unroll
        for (int r = 0; r < 8; ++r) {
            int row = r + 8 * hi;
            float v0 = s0[r] * (ATT_SCALE * msk[row * 32 + n]);
            float v1 = s1[r] * (ATT_SCALE * msk[row * 32 + 16 + n]);
            float vm = fmaxf(v0, v1);
            #pragma unroll
            for (int sh = 1; sh < 16; sh <<= 1) vm = fmaxf(vm, __shfl_xor(vm, sh, 32));
            float mnew = fmaxf(mrow[r], vm);
            float p0 = __expf(v0 - mnew);
            float p1 = __expf(v1 - mnew);
            float rs = p0 + p1;
            #pragma unroll
            for (int sh = 1; sh < 16; sh <<= 1) rs += __shfl_xor(rs, sh, 32);
            float al = __expf(mrow[r] - mnew);
            lrow[r] = lrow[r] * al + rs;
            mrow[r] = mnew;
            ps[wave][row * 32 + n]      = (__bf16)p0;
            ps[wave][row * 32 + 16 + n] = (__bf16)p1;
            o0[r] *= al; o1[r] *= al; o2[r] *= al; o3[r] *= al;
        }

        // reload P as A fragment (16 q-rows x 32 keys); same-wave DS is in-order
        v16bf pa;
        #pragma unroll
        for (int e = 0; e < 16; ++e) pa[e] = ps[wave][n * 32 + a_koff(e, hi)];

        // O += P @ V : four 16-wide d tiles
        {
            v16bf bm;
            #pragma unroll
            for (int e = 0; e < 16; ++e) bm[e] = vsw[b_koff(e, hi) * HD +  0 + n];
            o0 = __builtin_amdgcn_wmma_f32_16x16x32_bf16(false, pa, false, bm, (short)0, o0, false, false);
            #pragma unroll
            for (int e = 0; e < 16; ++e) bm[e] = vsw[b_koff(e, hi) * HD + 16 + n];
            o1 = __builtin_amdgcn_wmma_f32_16x16x32_bf16(false, pa, false, bm, (short)0, o1, false, false);
            #pragma unroll
            for (int e = 0; e < 16; ++e) bm[e] = vsw[b_koff(e, hi) * HD + 32 + n];
            o2 = __builtin_amdgcn_wmma_f32_16x16x32_bf16(false, pa, false, bm, (short)0, o2, false, false);
            #pragma unroll
            for (int e = 0; e < 16; ++e) bm[e] = vsw[b_koff(e, hi) * HD + 48 + n];
            o3 = __builtin_amdgcn_wmma_f32_16x16x32_bf16(false, pa, false, bm, (short)0, o3, false, false);
        }
        __syncthreads();   // protect shared msk before next iteration
#if HAVE_TDM
        buf ^= 1;
#endif
    }

    // normalize and write ctx rows ([B,S,E], columns h*64 .. h*64+63)
    #pragma unroll
    for (int r = 0; r < 8; ++r) {
        float inv = 1.0f / lrow[r];
        int s = q0 + r + 8 * hi;
        bf16_t* crow = ctx + ((size_t)b * SEQ + s) * EMBED + h * HD;
        crow[ 0 + n] = (__bf16)(o0[r] * inv);
        crow[16 + n] = (__bf16)(o1[r] * inv);
        crow[32 + n] = (__bf16)(o2[r] * inv);
        crow[48 + n] = (__bf16)(o3[r] * inv);
    }
}

// ---------------------------------------------------------------------------
// Kernel 3: output projection.  out[m,f] = sum_e ctx[m,e]*Wo[f,e] + bo[f]
// ---------------------------------------------------------------------------
__global__ void __launch_bounds__(256)
out_proj_kernel(const bf16_t* __restrict__ ctx, const bf16_t* __restrict__ Wob,
                const float* __restrict__ bo, float* __restrict__ out)
{
    __shared__ __align__(16) bf16_t cs[16 * EMBED];
    const int tid  = threadIdx.x;
    const int lane = tid & 31, wave = tid >> 5;
    const int n = lane & 15, hi = lane >> 4;
    const int mt = blockIdx.x;
    const int nt = blockIdx.y * 8 + wave;

    const uint4* src = (const uint4*)(ctx + (size_t)mt * 16 * EMBED);
    uint4* dst = (uint4*)cs;
    #pragma unroll
    for (int i = 0; i < 4; ++i) dst[tid + 256 * i] = src[tid + 256 * i];
    __syncthreads();

    const int f0 = nt * 16;
    const bf16_t* wrow = Wob + (size_t)(f0 + n) * EMBED;
    v8f acc = {};
    #pragma unroll
    for (int e0 = 0; e0 < EMBED; e0 += 32) {
        v16bf a, bm;
        #pragma unroll
        for (int e = 0; e < 16; ++e) {
            a[e]  = cs[n * EMBED + e0 + a_koff(e, hi)];
            bm[e] = wrow[e0 + b_koff(e, hi)];
        }
        acc = __builtin_amdgcn_wmma_f32_16x16x32_bf16(false, a, false, bm,
                                                      (short)0, acc, false, false);
    }
    const float bia = bo[f0 + n];
    #pragma unroll
    for (int r = 0; r < 8; ++r) {
        int m = mt * 16 + r + 8 * hi;
        out[(size_t)m * EMBED + f0 + n] = acc[r] + bia;
    }
}

// ---------------------------------------------------------------------------
extern "C" void kernel_launch(void* const* d_in, const int* in_sizes, int n_in,
                              void* d_out, int out_size, void* d_ws, size_t ws_size,
                              hipStream_t stream)
{
    const float* x    = (const float*)d_in[0];
    const float* adj  = (const float*)d_in[1];
    const float* bond = (const float*)d_in[2];
    const float* Wq = (const float*)d_in[3];  const float* bq = (const float*)d_in[4];
    const float* Wk = (const float*)d_in[5];  const float* bk = (const float*)d_in[6];
    const float* Wv = (const float*)d_in[7];  const float* bv = (const float*)d_in[8];
    const float* Wo = (const float*)d_in[9];  const float* bo = (const float*)d_in[10];
    float* out = (float*)d_out;

    const size_t QKV = (size_t)NB * HEADS * SEQ * HD;      // 8 Mi elems each
    bf16_t* Q   = (bf16_t*)d_ws;
    bf16_t* K   = Q + QKV;
    bf16_t* V   = K + QKV;
    bf16_t* CTX = V + QKV;
    bf16_t* WB  = CTX + QKV;          // 4 x 512 x 512 bf16 (Wq,Wk,Wv,Wo)

    wcvt_kernel<<<dim3(1024), 256, 0, stream>>>(Wq, Wk, Wv, Wo, WB);
    qkv_proj_kernel<<<dim3(NB * SEQ / 16, 4, 3), 256, 0, stream>>>(
        x, WB, bq, bk, bv, Q, K, V);
    attn_kernel<<<dim3(SEQ / 16, NB), 256, 0, stream>>>(
        Q, K, V, adj, bond, CTX);
    out_proj_kernel<<<dim3(NB * SEQ / 16, 4), 256, 0, stream>>>(
        CTX, Wo == nullptr ? WB : WB + 3 * 262144, bo, out);
}
